// Attention_67594195304671
// MI455X (gfx1250) — matrix operand order repaired and checked by reading, not compiled, not run on hardware
//
#include <hip/hip_runtime.h>
#include <hip/hip_bf16.h>
#include <stdint.h>

#define DIMX 768
#define NH   12
#define HD   64
#define BB   8
#define TT   1024
#define SCALEF 0.125f   // 64^-0.5

typedef __attribute__((ext_vector_type(16))) _Float16 v16h;
typedef __attribute__((ext_vector_type(8)))  float    v8f;

// ---- WMMA fragment loaders (layouts per CDNA5 ISA 7.12.2, wave32) ----
// A-fragment 16x32 f16: lane half h=lane>>4, row r=lane&15.
//   element e in [0,8):  K = h*8 + e
//   element e in [8,16): K = 16 + h*8 + (e-8)
__device__ __forceinline__ v16h load_a_f32(const float* __restrict__ p, int ld, int lane) {
  const int h = (lane >> 4) & 1, r = lane & 15;
  const float* row = p + r * ld + h * 8;
  v16h f;
#pragma unroll
  for (int i = 0; i < 8; ++i) f[i] = (_Float16)row[i];
#pragma unroll
  for (int i = 0; i < 8; ++i) f[8 + i] = (_Float16)row[16 + i];
  return f;
}
__device__ __forceinline__ v16h load_a_f16(const _Float16* __restrict__ p, int ld, int lane) {
  const int h = (lane >> 4) & 1, r = lane & 15;
  const _Float16* row = p + r * ld + h * 8;
  v16h f;
#pragma unroll
  for (int i = 0; i < 8; ++i) f[i] = row[i];
#pragma unroll
  for (int i = 0; i < 8; ++i) f[8 + i] = row[16 + i];
  return f;
}
// B-fragment 32x16 f16: lane half h, col n=lane&15. element e<8: K = h*8 + e; e>=8: K = 16 + h*8 + (e-8)
__device__ __forceinline__ v16h load_b_f16(const _Float16* __restrict__ p, int ld, int lane) {
  const int h = (lane >> 4) & 1, n = lane & 15;
  const _Float16* row = p + n * ld + h * 8;
  v16h f;
#pragma unroll
  for (int i = 0; i < 8; ++i) f[i] = row[i];
#pragma unroll
  for (int i = 0; i < 8; ++i) f[8 + i] = row[16 + i];
  return f;
}
// B-fragment sourced from an LDS-staged f32 tile laid out [64 cols][32 k] row-major
__device__ __forceinline__ v16h load_b_lds(const float* p, int lane) {
  const int h = (lane >> 4) & 1, n = lane & 15;
  const float* row = p + n * 32 + h * 8;
  v16h f;
#pragma unroll
  for (int i = 0; i < 8; ++i) f[i] = (_Float16)row[i];
#pragma unroll
  for (int i = 0; i < 8; ++i) f[8 + i] = (_Float16)row[16 + i];
  return f;
}
__device__ __forceinline__ v8f wmma16(v16h a, v16h b, v8f c) {
  return __builtin_amdgcn_wmma_f32_16x16x32_f16(false, a, false, b, (short)0, c, false, false);
}

// ---- CDNA5 async copy: 16B global -> LDS, tracked by ASYNCcnt ----
__device__ __forceinline__ void async_copy_b128(uint32_t lds_byte_addr, const float* gaddr) {
  asm volatile("global_load_async_to_lds_b128 %0, %1, off"
               :: "v"(lds_byte_addr), "v"(gaddr) : "memory");
}
// Stage one 64x32 f32 B tile (8KB) with 128 threads: 4 async b128 per lane.
__device__ __forceinline__ void stage_b_tile(float* Bs, const float* __restrict__ w,
                                             int n0, int k0, int tid) {
  const int row  = tid >> 1;            // 0..63  (weight row / output col)
  const int colf = (tid & 1) * 16;      // float offset within the 32-wide k slice
  const float* g = w + (size_t)(n0 + row) * DIMX + k0 + colf;
  const uint32_t l = (uint32_t)(uint64_t)(uintptr_t)(Bs + row * 32 + colf);
#pragma unroll
  for (int i = 0; i < 4; ++i)
    async_copy_b128(l + 16u * i, g + 4 * i);
}

// ================= Kernel 1: QKV projection =================
// C[8192,2304] = x[8192,768] @ qkv_w[2304,768]^T + qkv_b
// scatter -> Q[B,H,T,hd] f16, K[B,H,T,hd] f16, VT[B,H,hd,T] f16
__global__ __launch_bounds__(128) void qkv_kernel(
    const float* __restrict__ x, const float* __restrict__ w,
    const float* __restrict__ bias,
    _Float16* __restrict__ Qb, _Float16* __restrict__ Kb, _Float16* __restrict__ VT) {
  __shared__ float Bs[2][64 * 32];      // double-buffered shared B tile (16KB)
  const int tid  = threadIdx.x;
  const int lane = tid & 31;
  const int wave = tid >> 5;
  const int m0 = blockIdx.x * 64 + wave * 16;   // 128 M-tiles
  const int n0 = blockIdx.y * 64;               // 36 N-tiles (shared by all 4 waves)

  stage_b_tile(Bs[0], w, n0, 0, tid);           // prologue stage

  v8f acc[4] = {};
  const float* xrow = x + (size_t)m0 * DIMX;
  for (int ks = 0; ks < 24; ++ks) {             // K = 768 in steps of 32
    const int k0 = ks * 32;
    const float* cur = Bs[ks & 1];
    if (ks + 1 < 24) {
      stage_b_tile(Bs[(ks + 1) & 1], w, n0, k0 + 32, tid);
      asm volatile("s_wait_asynccnt 0x4" ::: "memory");  // prev stage done (in-order)
    } else {
      asm volatile("s_wait_asynccnt 0x0" ::: "memory");
    }
    __syncthreads();
    v16h a = load_a_f32(xrow + k0, DIMX, lane);
#pragma unroll
    for (int nt = 0; nt < 4; ++nt)
      acc[nt] = wmma16(a, load_b_lds(cur + nt * 16 * 32, lane), acc[nt]);
    __syncthreads();                            // all waves done with cur buffer
  }

  const int half = lane >> 4, nl = lane & 15;
#pragma unroll
  for (int nt = 0; nt < 4; ++nt) {
    const int n = n0 + nt * 16 + nl;
    const float bv = bias[n];
    const int sec = n / DIMX;          // 0=Q 1=K 2=V
    const int d = n % DIMX;
    const int h = d >> 6, hd = d & 63;
#pragma unroll
    for (int j = 0; j < 8; ++j) {
      const int m = m0 + j + half * 8;
      const int b = m >> 10, t = m & 1023;
      const float v = acc[nt][j] + bv;
      const size_t bh = (size_t)(b * NH + h);
      if (sec == 0)      Qb[(bh * TT + t) * HD + hd] = (_Float16)v;
      else if (sec == 1) Kb[(bh * TT + t) * HD + hd] = (_Float16)v;
      else               VT[(bh * HD + hd) * TT + t] = (_Float16)v;
    }
  }
}

// ================= Kernel 2: attention core =================
// One wave per 16-query block. Full 16x1024 score strip in LDS (64KB/wave).
__global__ __launch_bounds__(128) void attn_kernel(
    const _Float16* __restrict__ Qb, const _Float16* __restrict__ Kb,
    const _Float16* __restrict__ VT, _Float16* __restrict__ AO) {
  extern __shared__ float smem[];
  const int lane = threadIdx.x & 31;
  const int wave = threadIdx.x >> 5;
  float* S = smem + wave * (16 * TT);

  const int gb = blockIdx.x * 4 + wave;   // 0..6143
  const int bh = gb >> 6;                 // b*12+h (same for all 4 waves in block)
  const int qi = gb & 63;                 // query tile within T
  const _Float16* Qp = Qb + ((size_t)bh * TT + qi * 16) * HD;
  const _Float16* Kp = Kb + (size_t)bh * TT * HD;
  const _Float16* Vp = VT + (size_t)bh * HD * TT;
  const int half = lane >> 4, nl = lane & 15;

  // ---- Phase 1: S = scale * Q @ K^T  (16 x 1024) -> LDS ----
  v16h a0 = load_a_f16(Qp, HD, lane);        // K-dim 0..31
  v16h a1 = load_a_f16(Qp + 32, HD, lane);   // K-dim 32..63
  for (int kt = 0; kt < 64; ++kt) {
    v16h b0 = load_b_f16(Kp + (size_t)kt * 16 * HD, HD, lane);
    v16h b1 = load_b_f16(Kp + (size_t)kt * 16 * HD + 32, HD, lane);
    v8f c = {};
    c = wmma16(a0, b0, c);
    c = wmma16(a1, b1, c);
#pragma unroll
    for (int j = 0; j < 8; ++j)
      S[(j + half * 8) * TT + kt * 16 + nl] = c[j] * SCALEF;
  }
  __syncthreads();

  // ---- Phase 2: row max + row sum of exp (half row per lane) ----
  const int r = nl;
  const float4* Srow = (const float4*)(S + r * TT + half * 512);
  float m = -1e30f;
#pragma unroll 4
  for (int i = 0; i < 128; ++i) {
    float4 v = Srow[i];
    m = fmaxf(m, fmaxf(fmaxf(v.x, v.y), fmaxf(v.z, v.w)));
  }
  m = fmaxf(m, __shfl_xor(m, 16));
  float l = 0.f;
#pragma unroll 4
  for (int i = 0; i < 128; ++i) {
    float4 v = Srow[i];
    l += __expf(v.x - m) + __expf(v.y - m) + __expf(v.z - m) + __expf(v.w - m);
  }
  l += __shfl_xor(l, 16);
  const float rl = 1.0f / l;
  __syncthreads();

  // ---- Phase 3: O = softmax(S) @ V  via on-the-fly exp A-fragments ----
  v8f o[4] = {};
  for (int kc = 0; kc < 32; ++kc) {
    const float* sp = S + r * TT + kc * 32 + half * 8;
    v16h a;
#pragma unroll
    for (int i = 0; i < 8; ++i) a[i] = (_Float16)__expf(sp[i] - m);
#pragma unroll
    for (int i = 0; i < 8; ++i) a[8 + i] = (_Float16)__expf(sp[16 + i] - m);
#pragma unroll
    for (int nt = 0; nt < 4; ++nt) {
      v16h bf = load_b_f16(Vp + (size_t)(nt * 16) * TT + kc * 32, TT, lane);
      o[nt] = wmma16(a, bf, o[nt]);
    }
  }

  // ---- normalize + write AO[B,T,D] f16 ----
  const int b = bh / NH, h = bh % NH;
#pragma unroll
  for (int nt = 0; nt < 4; ++nt) {
#pragma unroll
    for (int j = 0; j < 8; ++j) {
      const float sc = __shfl(rl, j + half * 8);   // 1/l of this C-row
      const int trow = qi * 16 + j + half * 8;
      AO[((size_t)b * TT + trow) * DIMX + h * 64 + nt * 16 + nl] =
          (_Float16)(o[nt][j] * sc);
    }
  }
}

// ================= Kernel 3: output projection =================
// out[8192,768] = AO[8192,768] @ proj_w[768,768]^T + proj_b   (f32 out)
__global__ __launch_bounds__(128) void proj_kernel(
    const _Float16* __restrict__ ain, const float* __restrict__ w,
    const float* __restrict__ bias, float* __restrict__ out) {
  __shared__ float Bs[2][64 * 32];
  const int tid  = threadIdx.x;
  const int lane = tid & 31;
  const int wave = tid >> 5;
  const int m0 = blockIdx.x * 64 + wave * 16;   // 128 M-tiles
  const int n0 = blockIdx.y * 64;               // 12 N-tiles

  stage_b_tile(Bs[0], w, n0, 0, tid);

  v8f acc[4] = {};
  const _Float16* arow = ain + (size_t)m0 * DIMX;
  for (int ks = 0; ks < 24; ++ks) {
    const int k0 = ks * 32;
    const float* cur = Bs[ks & 1];
    if (ks + 1 < 24) {
      stage_b_tile(Bs[(ks + 1) & 1], w, n0, k0 + 32, tid);
      asm volatile("s_wait_asynccnt 0x4" ::: "memory");
    } else {
      asm volatile("s_wait_asynccnt 0x0" ::: "memory");
    }
    __syncthreads();
    v16h a = load_a_f16(arow + k0, DIMX, lane);
#pragma unroll
    for (int nt = 0; nt < 4; ++nt)
      acc[nt] = wmma16(a, load_b_lds(cur + nt * 16 * 32, lane), acc[nt]);
    __syncthreads();
  }

  const int half = lane >> 4, nl = lane & 15;
#pragma unroll
  for (int nt = 0; nt < 4; ++nt) {
    const int n = n0 + nt * 16 + nl;
    const float bv = bias[n];
#pragma unroll
    for (int j = 0; j < 8; ++j) {
      const int m = m0 + j + half * 8;
      out[(size_t)m * DIMX + n] = acc[nt][j] + bv;
    }
  }
}

extern "C" void kernel_launch(void* const* d_in, const int* in_sizes, int n_in,
                              void* d_out, int out_size, void* d_ws, size_t ws_size,
                              hipStream_t stream) {
  const float* x      = (const float*)d_in[0];
  const float* qkv_w  = (const float*)d_in[1];
  const float* qkv_b  = (const float*)d_in[2];
  const float* proj_w = (const float*)d_in[3];
  const float* proj_b = (const float*)d_in[4];
  float* out = (float*)d_out;

  const size_t seg = (size_t)BB * NH * TT * HD * sizeof(_Float16); // 12.6 MB
  char* ws = (char*)d_ws;
  _Float16* Qb = (_Float16*)(ws);
  _Float16* Kb = (_Float16*)(ws + seg);
  _Float16* VT = (_Float16*)(ws + 2 * seg);
  _Float16* AO = (_Float16*)(ws + 3 * seg);

  // QKV projection: M=8192 (128 tiles), N=2304 (36 tiles)
  qkv_kernel<<<dim3(128, 36), 128, 0, stream>>>(x, qkv_w, qkv_b, Qb, Kb, VT);

  // Attention: 8*12*64 = 6144 query blocks, 4 waves/block -> 1536 blocks
  // Dynamic LDS: 4 waves * 16*1024 floats = 256 KB (of 320 KB/WGP)
  attn_kernel<<<1536, 128, 4 * 16 * TT * sizeof(float), stream>>>(Qb, Kb, VT, AO);

  // Output projection: M=8192 (128 tiles), N=768 (12 tiles)
  proj_kernel<<<dim3(128, 12), 128, 0, stream>>>(AO, proj_w, proj_b, out);
}
